// att_hgraph_learner_54013508714671
// MI455X (gfx1250) — compile-verified, matrix operand order checked by reading, not run on hardware
//
#include <hip/hip_runtime.h>

// ---------------------------------------------------------------------------
// CDNA5 (gfx1250) attention-hypergraph GPS.
// - All big GEMMs: v_wmma_f32_16x16x32_bf16, fp32 accumulate.
// - Every operand converted to bf16 exactly once (weights pre-transposed to
//   [N][K]); GEMM staging is a pure copy -> async global->LDS when available.
// - LN / softmax / top-k stay fp32 in VALU.
// ---------------------------------------------------------------------------

#if defined(__has_builtin)
#  if __has_builtin(__builtin_amdgcn_global_load_async_to_lds_b64) && \
      __has_builtin(__builtin_amdgcn_s_wait_asynccnt)
#    define USE_ASYNC_LDS 1
#  endif
#endif
#ifndef USE_ASYNC_LDS
#  define USE_ASYNC_LDS 0
#endif

typedef __attribute__((ext_vector_type(16))) __bf16 bf16x16;
typedef __attribute__((ext_vector_type(8)))  float  f32x8;
typedef __attribute__((ext_vector_type(2)))  int    v2i;

#if USE_ASYNC_LDS
typedef __attribute__((address_space(1))) v2i* gv2i_p;   // global 64-bit chunk
typedef __attribute__((address_space(3))) v2i* lv2i_p;   // LDS 64-bit chunk
#endif

#define BATCH 8
#define NCH   64
#define NNODE 1024
#define HID   384
#define NLAY  3
#define KNN   16
#define NHEAD 3
#define DHEAD 128
#define NOUT  128

// ---- bf16 <-> f32 helpers (storage-only __bf16) ----------------------------
__device__ __forceinline__ __bf16 f2bf(float x) {
  union { float f; unsigned u; } c; c.f = x;
  unsigned r = (c.u + 0x7fffu + ((c.u >> 16) & 1u)) >> 16;
  unsigned short s = (unsigned short)r;
  return *reinterpret_cast<__bf16*>(&s);
}
__device__ __forceinline__ float bf2f(__bf16 x) {
  unsigned short s = *reinterpret_cast<unsigned short*>(&x);
  union { unsigned u; float f; } c; c.u = ((unsigned)s) << 16;
  return c.f;
}
__device__ __forceinline__ f32x8 zero8() {
  f32x8 z;
#pragma unroll
  for (int i = 0; i < 8; ++i) z[i] = 0.f;
  return z;
}
__device__ __forceinline__ f32x8 wmma_bf16(bf16x16 a, bf16x16 b, f32x8 c) {
  return __builtin_amdgcn_wmma_f32_16x16x32_bf16(false, a, false, b, (short)0, c,
                                                 false, false);
}

// ---- global->LDS staging of 4 bf16 (8 bytes) -------------------------------
__device__ __forceinline__ void stage4(const __bf16* __restrict__ g, __bf16* l) {
#if USE_ASYNC_LDS
  __builtin_amdgcn_global_load_async_to_lds_b64((gv2i_p)(void*)g, (lv2i_p)(void*)l,
                                                0, 0);
#else
  *reinterpret_cast<unsigned long long*>(l) =
      *reinterpret_cast<const unsigned long long*>(g);
#endif
}
__device__ __forceinline__ void stage_fence() {
#if USE_ASYNC_LDS
  __builtin_amdgcn_s_wait_asynccnt(0);
#endif
}

// K index inside a 16x32 bf16 fragment: VGPR v, lane-half h.
__device__ __forceinline__ int frag_kbase(int v, int half) {
  return ((v >> 2) << 4) + (half << 3) + ((v & 3) << 1);
}

// Load a 16x32 fragment from bf16 memory: element (m,k) at base + m*sm + k*sk.
__device__ __forceinline__ bf16x16 load_frag_bf16(const __bf16* base, int sm, int sk) {
  const int lane = threadIdx.x & 31;
  const int m = lane & 15, half = lane >> 4;
  const __bf16* p = base + m * sm;
  bf16x16 f;
#pragma unroll
  for (int v = 0; v < 8; ++v) {
    int kb = frag_kbase(v, half);
    f[2 * v]     = p[kb * sk];
    f[2 * v + 1] = p[(kb + 1) * sk];
  }
  return f;
}

// Store a 16x16 f32 D tile (+bias, +relu); f32 and/or bf16 destinations.
__device__ __forceinline__ void store_tile(float* C, __bf16* Cb, int ldc, f32x8 acc,
                                           const float* biasN, int relu) {
  const int lane = threadIdx.x & 31;
  const int n = lane & 15, mh = lane >> 4;
#pragma unroll
  for (int r = 0; r < 8; ++r) {
    int m = r + (mh << 3);
    float v = acc[r];
    if (biasN) v += biasN[n];
    if (relu) v = fmaxf(v, 0.f);
    if (C)  C[(long long)m * ldc + n] = v;
    if (Cb) Cb[(long long)m * ldc + n] = f2bf(v);
  }
}

// ---------------------------------------------------------------------------
// Batched GEMM, all-bf16 operands:  C[b] = act(A[b] @ Bt^T + bias)
//   A : bf16 row-major [M][K], batch stride sAb
//   Bt: bf16 pre-transposed weights [N][K]  (shared across batch)
// WG tile 64(M) x 128(N), K-step 32, 8 waves each owning a 32x32 block.
// Staging is a pure bf16 copy (async global->LDS when the builtin exists).
// ---------------------------------------------------------------------------
__global__ __launch_bounds__(256)
void gemm_bf16_wmma(const __bf16* __restrict__ A, long long sAb,
                    const __bf16* __restrict__ Bt,
                    const float* __restrict__ bias,
                    float* __restrict__ C, __bf16* __restrict__ Cb, long long sCb,
                    int M, int N, int K, int relu) {
  const int b = blockIdx.z;
  A += (long long)b * sAb;
  if (C)  C  += (long long)b * sCb;
  if (Cb) Cb += (long long)b * sCb;

  const int row0 = blockIdx.y * 64;
  const int col0 = blockIdx.x * 128;

  // padded row stride 36 elems (72B, 8B-aligned rows for b64 staging)
  __shared__ __align__(16) __bf16 As[64 * 36];
  __shared__ __align__(16) __bf16 Bs[128 * 36];

  const int tid = threadIdx.x;
  const int wave = tid >> 5;
  const int wr = wave >> 2;        // 0..1
  const int wc = wave & 3;         // 0..3

  f32x8 acc00 = zero8(), acc01 = zero8(), acc10 = zero8(), acc11 = zero8();

  for (int k0 = 0; k0 < K; k0 += 32) {
    __syncthreads();
#pragma unroll
    for (int i = 0; i < 2; ++i) {            // A: 64x32 = 512 chunks of 4 bf16
      int c = tid + i * 256;
      int row = c >> 3, seg = c & 7;
      stage4(A + (long long)(row0 + row) * K + k0 + seg * 4,
             &As[row * 36 + seg * 4]);
    }
#pragma unroll
    for (int i = 0; i < 4; ++i) {            // B: 128x32 = 1024 chunks of 4 bf16
      int c = tid + i * 256;
      int row = c >> 3, seg = c & 7;
      stage4(Bt + (long long)(col0 + row) * K + k0 + seg * 4,
             &Bs[row * 36 + seg * 4]);
    }
    if (k0 + 32 < K) {                       // gfx1250 global_prefetch_b8
      __builtin_prefetch(A + (long long)(row0 + tid % 64) * K + k0 + 32, 0, 1);
      __builtin_prefetch(Bt + (long long)(col0 + tid % 128) * K + k0 + 32, 0, 1);
    }
    stage_fence();                           // s_wait_asynccnt 0 (if async path)
    __syncthreads();

    bf16x16 a0 = load_frag_bf16(&As[(wr * 32) * 36], 36, 1);
    bf16x16 a1 = load_frag_bf16(&As[(wr * 32 + 16) * 36], 36, 1);
    bf16x16 b0 = load_frag_bf16(&Bs[(wc * 32) * 36], 36, 1);
    bf16x16 b1 = load_frag_bf16(&Bs[(wc * 32 + 16) * 36], 36, 1);
    acc00 = wmma_bf16(a0, b0, acc00);
    acc01 = wmma_bf16(a0, b1, acc01);
    acc10 = wmma_bf16(a1, b0, acc10);
    acc11 = wmma_bf16(a1, b1, acc11);
  }

  const int rt = row0 + wr * 32;
  const int ct = col0 + wc * 32;
  long long o00 = (long long)rt * N + ct;
  long long o10 = (long long)(rt + 16) * N + ct;
  store_tile(C ? C + o00      : nullptr, Cb ? Cb + o00      : nullptr, N, acc00,
             bias ? bias + ct      : nullptr, relu);
  store_tile(C ? C + o00 + 16 : nullptr, Cb ? Cb + o00 + 16 : nullptr, N, acc01,
             bias ? bias + ct + 16 : nullptr, relu);
  store_tile(C ? C + o10      : nullptr, Cb ? Cb + o10      : nullptr, N, acc10,
             bias ? bias + ct      : nullptr, relu);
  store_tile(C ? C + o10 + 16 : nullptr, Cb ? Cb + o10 + 16 : nullptr, N, acc11,
             bias ? bias + ct + 16 : nullptr, relu);
}

// ---------------------------------------------------------------------------
// One-time conversions: weights f32[K][N] -> bf16 transposed [N][K]; x -> bf16
// ---------------------------------------------------------------------------
__global__ void cvt_w_kernel(const float* __restrict__ W, __bf16* __restrict__ Wt,
                             int K, int N) {
  const long long mat = (long long)blockIdx.y * K * N;
  int idx = blockIdx.x * 256 + threadIdx.x;
  if (idx < K * N) {
    int k = idx / N, n = idx - k * N;
    Wt[mat + (long long)n * K + k] = f2bf(W[mat + idx]);
  }
}

__global__ void cvt_x_kernel(const float* __restrict__ x, __bf16* __restrict__ xb) {
  int idx = blockIdx.x * 256 + threadIdx.x;     // over B*N*C
  if (idx < BATCH * NNODE * NCH) {
    int b = idx / (NNODE * NCH);
    int r = idx - b * NNODE * NCH;
    int n = r / NCH, c = r - n * NCH;
    xb[idx] = f2bf(x[(long long)b * NCH * NNODE + (long long)c * NNODE + n]);
  }
}

// ---------------------------------------------------------------------------
// Per-node L2 normalization -> bf16 (feeds cosine-sim WMMA).
// ---------------------------------------------------------------------------
__global__ void normalize_kernel(const float* __restrict__ h, __bf16* __restrict__ nh) {
  const int b = blockIdx.y, node = blockIdx.x;
  const long long base = ((long long)(b * NNODE + node)) * HID;
  const int tid = threadIdx.x;                  // 128 threads
  float s = 0.f;
  for (int i = tid; i < HID; i += 128) { float v = h[base + i]; s += v * v; }
  __shared__ float red[128];
  red[tid] = s; __syncthreads();
  for (int off = 64; off; off >>= 1) {
    if (tid < off) red[tid] += red[tid + off];
    __syncthreads();
  }
  const float rn = rsqrtf(red[0] + 1e-30f);
  for (int i = tid; i < HID; i += 128) nh[base + i] = f2bf(h[base + i] * rn);
}

// ---------------------------------------------------------------------------
// Fused cosine-sim (WMMA, nh @ nh^T) + top-16 per row. 16 rows per WG.
// ---------------------------------------------------------------------------
__global__ __launch_bounds__(256)
void simtopk_kernel(const __bf16* __restrict__ nh, int* __restrict__ knn) {
  const int b = blockIdx.y;
  const int row0 = blockIdx.x * 16;
  const int tid = threadIdx.x;
  const int wave = tid >> 5;
  const int lane = tid & 31;

  __shared__ __bf16 SC[16][1024];
  __shared__ float rv[8];
  __shared__ int   ri[8];

  f32x8 acc[8];
#pragma unroll
  for (int t = 0; t < 8; ++t) acc[t] = zero8();

  const __bf16* arow = nh + ((long long)(b * NNODE + row0)) * HID;
  for (int ks = 0; ks < HID / 32; ++ks) {
    bf16x16 af = load_frag_bf16(arow + ks * 32, HID, 1);
#pragma unroll
    for (int t = 0; t < 8; ++t) {
      const __bf16* brow = nh + ((long long)(b * NNODE + wave * 128 + t * 16)) * HID + ks * 32;
      bf16x16 bf = load_frag_bf16(brow, HID, 1);
      acc[t] = wmma_bf16(af, bf, acc[t]);
    }
  }
  {
    const int n = lane & 15, mh = lane >> 4;
#pragma unroll
    for (int t = 0; t < 8; ++t)
#pragma unroll
      for (int r = 0; r < 8; ++r) {
        int m = r + (mh << 3);
        int c = wave * 128 + t * 16 + n;
        float v = acc[t][r];
        if (c == row0 + m) v -= 2.0f;          // exclude self
        SC[m][c] = f2bf(v);
      }
  }
  __syncthreads();

  for (int r = 0; r < 16; ++r) {
    for (int kk = 0; kk < KNN; ++kk) {
      float bvv = -1e30f; int bii = 0;
#pragma unroll
      for (int i = 0; i < 4; ++i) {
        int c = tid + i * 256;
        float v = bf2f(SC[r][c]);
        if (v > bvv) { bvv = v; bii = c; }
      }
      for (int off = 16; off; off >>= 1) {
        float ov = __shfl_xor(bvv, off, 32);
        int   oi = __shfl_xor(bii, off, 32);
        if (ov > bvv || (ov == bvv && oi < bii)) { bvv = ov; bii = oi; }
      }
      if (lane == 0) { rv[wave] = bvv; ri[wave] = bii; }
      __syncthreads();
      if (tid == 0) {
        float bv2 = rv[0]; int bi2 = ri[0];
        for (int w = 1; w < 8; ++w)
          if (rv[w] > bv2 || (rv[w] == bv2 && ri[w] < bi2)) { bv2 = rv[w]; bi2 = ri[w]; }
        knn[((long long)(b * NNODE + row0 + r)) * KNN + kk] = bi2;
        SC[r][bi2] = f2bf(-1e30f);
      }
      __syncthreads();
    }
  }
}

// ---------------------------------------------------------------------------
// Fused multihead attention (bf16 Q/K/V in, bf16 O out):
// WMMA scores (scale folded into P store) -> LDS softmax -> WMMA P@V.
// ---------------------------------------------------------------------------
__global__ __launch_bounds__(256)
void attn_kernel(const __bf16* __restrict__ Q, const __bf16* __restrict__ Km,
                 const __bf16* __restrict__ V, __bf16* __restrict__ O) {
  const int b = blockIdx.z, head = blockIdx.y;
  const int row0 = blockIdx.x * 16;
  const int hc = head * DHEAD;
  const int tid = threadIdx.x;
  const int wave = tid >> 5;
  const int lane = tid & 31;

  __shared__ __bf16 P[16][1024];
  __shared__ float red[16];
  __shared__ float rowinv[16];

  const float qscale = 0.08838834764831845f;   // 1/sqrt(128)

  // --- phase 1: scores S = Q @ K^T ; wave owns 128 key columns ---
  f32x8 acc[8];
#pragma unroll
  for (int t = 0; t < 8; ++t) acc[t] = zero8();
  const __bf16* qb = Q + ((long long)(b * NNODE + row0)) * HID + hc;
  for (int ks = 0; ks < DHEAD / 32; ++ks) {
    bf16x16 af = load_frag_bf16(qb + ks * 32, HID, 1);
#pragma unroll
    for (int t = 0; t < 8; ++t) {
      const __bf16* kp = Km + ((long long)(b * NNODE + wave * 128 + t * 16)) * HID + hc + ks * 32;
      bf16x16 bf = load_frag_bf16(kp, HID, 1);
      acc[t] = wmma_bf16(af, bf, acc[t]);
    }
  }
  {
    const int n = lane & 15, mh = lane >> 4;
#pragma unroll
    for (int t = 0; t < 8; ++t)
#pragma unroll
      for (int r = 0; r < 8; ++r)
        P[r + (mh << 3)][wave * 128 + t * 16 + n] = f2bf(acc[t][r] * qscale);
  }
  __syncthreads();

  // --- phase 2: row softmax (max, exp, sum; 1/sum folded into epilogue) ---
  for (int r = 0; r < 16; ++r) {
    float lm = -1e30f;
#pragma unroll
    for (int i = 0; i < 4; ++i) lm = fmaxf(lm, bf2f(P[r][tid + i * 256]));
    for (int off = 16; off; off >>= 1) lm = fmaxf(lm, __shfl_xor(lm, off, 32));
    if (lane == 0) red[wave] = lm;
    __syncthreads();
    float rowmax = red[0];
    for (int w = 1; w < 8; ++w) rowmax = fmaxf(rowmax, red[w]);
    float ls = 0.f;
#pragma unroll
    for (int i = 0; i < 4; ++i) {
      int c = tid + i * 256;
      float e = __expf(bf2f(P[r][c]) - rowmax);
      P[r][c] = f2bf(e);
      ls += e;
    }
    for (int off = 16; off; off >>= 1) ls += __shfl_xor(ls, off, 32);
    if (lane == 0) red[8 + wave] = ls;
    __syncthreads();
    if (tid == 0) {
      float s = 0.f;
      for (int w = 0; w < 8; ++w) s += red[8 + w];
      rowinv[r] = 1.0f / s;
    }
  }
  __syncthreads();

  // --- phase 3: O = P @ V ; wave owns one 16x16 output tile ---
  f32x8 oacc = zero8();
  for (int ks = 0; ks < NNODE / 32; ++ks) {
    bf16x16 af = load_frag_bf16(&P[0][ks * 32], 1024, 1);
    const __bf16* vb = V + ((long long)(b * NNODE + ks * 32)) * HID + hc + wave * 16;
    bf16x16 bf = load_frag_bf16(vb, 1, HID);   // element (k,n): n stride 1, k stride HID
    oacc = wmma_bf16(af, bf, oacc);
  }
  {
    const int n = lane & 15, mh = lane >> 4;
    __bf16* ob = O + ((long long)(b * NNODE + row0)) * HID + hc + wave * 16;
#pragma unroll
    for (int r = 0; r < 8; ++r) {
      int m = r + (mh << 3);
      ob[(long long)m * HID + n] = f2bf(oacc[r] * rowinv[m]);
    }
  }
}

// ---------------------------------------------------------------------------
// GCN aggregate (deg==17 everywhere) + residual + LayerNorm. Wave per node.
// ---------------------------------------------------------------------------
__global__ __launch_bounds__(256)
void gcn_ln_kernel(const float* __restrict__ h, const float* __restrict__ xw,
                   const int* __restrict__ knn,
                   const float* __restrict__ lsc, const float* __restrict__ lbi,
                   float* __restrict__ out) {
  const int b = blockIdx.y;
  const int node = blockIdx.x * 8 + (threadIdx.x >> 5);
  const int lane = threadIdx.x & 31;
  const long long base = ((long long)(b * NNODE + node)) * HID;
  const int* kn = knn + ((long long)(b * NNODE + node)) * KNN;

  float vals[12];
#pragma unroll
  for (int i = 0; i < 12; ++i) vals[i] = xw[base + lane + i * 32];
  for (int j = 0; j < KNN; ++j) {
    long long nb = ((long long)(b * NNODE + kn[j])) * HID;
#pragma unroll
    for (int i = 0; i < 12; ++i) vals[i] += xw[nb + lane + i * 32];
  }
  const float inv17 = 1.0f / 17.0f;
  float mu = 0.f;
#pragma unroll
  for (int i = 0; i < 12; ++i) {
    vals[i] = h[base + lane + i * 32] + vals[i] * inv17;
    mu += vals[i];
  }
  for (int off = 16; off; off >>= 1) mu += __shfl_xor(mu, off, 32);
  mu *= (1.0f / HID);
  float var = 0.f;
#pragma unroll
  for (int i = 0; i < 12; ++i) { float d = vals[i] - mu; var += d * d; }
  for (int off = 16; off; off >>= 1) var += __shfl_xor(var, off, 32);
  var *= (1.0f / HID);
  const float rs = rsqrtf(var + 1e-5f);
#pragma unroll
  for (int i = 0; i < 12; ++i) {
    int d = lane + i * 32;
    out[base + d] = (vals[i] - mu) * rs * lsc[d] + lbi[d];
  }
}

// LayerNorm of (a + r); optional bf16 duplicate of the output.
__global__ __launch_bounds__(256)
void ln_kernel(const float* __restrict__ a, const float* __restrict__ r,
               const float* __restrict__ lsc, const float* __restrict__ lbi,
               float* __restrict__ out, __bf16* __restrict__ outb) {
  const int b = blockIdx.y;
  const int node = blockIdx.x * 8 + (threadIdx.x >> 5);
  const int lane = threadIdx.x & 31;
  const long long base = ((long long)(b * NNODE + node)) * HID;
  float vals[12];
  float mu = 0.f;
#pragma unroll
  for (int i = 0; i < 12; ++i) {
    long long idx = base + lane + i * 32;
    vals[i] = a[idx] + (r ? r[idx] : 0.f);
    mu += vals[i];
  }
  for (int off = 16; off; off >>= 1) mu += __shfl_xor(mu, off, 32);
  mu *= (1.0f / HID);
  float var = 0.f;
#pragma unroll
  for (int i = 0; i < 12; ++i) { float d = vals[i] - mu; var += d * d; }
  for (int off = 16; off; off >>= 1) var += __shfl_xor(var, off, 32);
  var *= (1.0f / HID);
  const float rs = rsqrtf(var + 1e-5f);
#pragma unroll
  for (int i = 0; i < 12; ++i) {
    int d = lane + i * 32;
    float v = (vals[i] - mu) * rs * lsc[d] + lbi[d];
    out[base + d] = v;
    if (outb) outb[base + d] = f2bf(v);
  }
}

// out = a + b (f32) and bf16 duplicate (feeds the MLP GEMM).
__global__ void add_kernel(const float* __restrict__ a, const float* __restrict__ b,
                           float* __restrict__ out, __bf16* __restrict__ outb, int n) {
  int i = blockIdx.x * blockDim.x + threadIdx.x;
  if (i < n) {
    float v = a[i] + b[i];
    out[i] = v;
    outb[i] = f2bf(v);
  }
}

__global__ void pool_kernel(const float* __restrict__ h, float* __restrict__ pooled) {
  const int b = blockIdx.x, d = threadIdx.x;        // blockDim = 384
  float s = 0.f;
  for (int n = 0; n < NNODE; ++n) s += h[((long long)(b * NNODE + n)) * HID + d];
  pooled[b * HID + d] = s * (1.0f / NNODE);
}

__global__ void head1_kernel(const float* __restrict__ pooled,
                             const float* __restrict__ W, const float* __restrict__ bia,
                             float* __restrict__ z) {
  const int b = blockIdx.x, d = threadIdx.x;        // blockDim = 384
  float s = bia[d];
  for (int k = 0; k < HID; ++k) s += pooled[b * HID + k] * W[k * HID + d];
  z[b * HID + d] = fmaxf(s, 0.f);
}

__global__ void head2_kernel(const float* __restrict__ z,
                             const float* __restrict__ W, const float* __restrict__ bia,
                             float* __restrict__ out) {
  const int b = blockIdx.x, d = threadIdx.x;        // blockDim = 128
  float s = bia[d];
  for (int k = 0; k < HID; ++k) s += z[b * HID + k] * W[k * NOUT + d];
  out[b * NOUT + d] = s;
}

// ---------------------------------------------------------------------------
// Host orchestration
// ---------------------------------------------------------------------------
extern "C" void kernel_launch(void* const* d_in, const int* in_sizes, int n_in,
                              void* d_out, int out_size, void* d_ws, size_t ws_size,
                              hipStream_t stream) {
  const float* x       = (const float*)d_in[0];
  const float* W_enc   = (const float*)d_in[1];
  const float* b_enc   = (const float*)d_in[2];
  const float* gcn_W   = (const float*)d_in[3];
  const float* gcn_b   = (const float*)d_in[4];
  const float* Wq      = (const float*)d_in[5];
  const float* bq      = (const float*)d_in[6];
  const float* Wk      = (const float*)d_in[7];
  const float* bk      = (const float*)d_in[8];
  const float* Wv      = (const float*)d_in[9];
  const float* bv      = (const float*)d_in[10];
  const float* Wo      = (const float*)d_in[11];
  const float* bo      = (const float*)d_in[12];
  const float* ln_sc   = (const float*)d_in[13];
  const float* ln_bi   = (const float*)d_in[14];
  const float* mlp_W1  = (const float*)d_in[15];
  const float* mlp_b1  = (const float*)d_in[16];
  const float* mlp_W2  = (const float*)d_in[17];
  const float* mlp_b2  = (const float*)d_in[18];
  const float* lin1_W  = (const float*)d_in[19];
  const float* lin1_b  = (const float*)d_in[20];
  const float* lin2_W  = (const float*)d_in[21];
  const float* lin2_b  = (const float*)d_in[22];
  (void)in_sizes; (void)n_in; (void)out_size; (void)ws_size;

  const long long SZ  = (long long)BATCH * NNODE * HID;   // 3,145,728 elements
  const long long HBN = (long long)NNODE * HID;           // activation batch stride
  const long long H2  = (long long)NNODE * 2 * HID;       // t1 batch stride

  // workspace carving (256B aligned)
  char* wp = (char*)d_ws;
  auto alloc = [&](long long bytes) -> void* {
    void* r = (void*)wp;
    wp += (bytes + 255) & ~255ll;
    return r;
  };
  float* buf_h   = (float*)alloc(SZ * 4);            // hcur f32
  float* buf_xw  = (float*)alloc(SZ * 4);            // xw / o-proj / mlp-t2
  float* buf_hl  = (float*)alloc(SZ * 4);            // h_local
  float* buf_ha  = (float*)alloc(SZ * 4);            // h_attn
  float* buf_cm  = (float*)alloc(SZ * 4);            // out_comb f32
  float* buf_pool= (float*)alloc(BATCH * HID * 4);
  float* buf_z   = (float*)alloc(BATCH * HID * 4);
  __bf16* buf_hb = (__bf16*)alloc(SZ * 2);           // hcur bf16
  __bf16* buf_nh = (__bf16*)alloc(SZ * 2);           // normalized h0 bf16
  __bf16* buf_qb = (__bf16*)alloc(SZ * 2);
  __bf16* buf_kb = (__bf16*)alloc(SZ * 2);
  __bf16* buf_vb = (__bf16*)alloc(SZ * 2);
  __bf16* buf_ob = (__bf16*)alloc(SZ * 2);           // attention out bf16
  __bf16* buf_cb = (__bf16*)alloc(SZ * 2);           // out_comb bf16
  __bf16* buf_t1 = (__bf16*)alloc(2 * SZ * 2);       // MLP hidden bf16
  __bf16* buf_xb = (__bf16*)alloc((long long)BATCH * NNODE * NCH * 2);
  __bf16* wt_enc = (__bf16*)alloc((long long)NCH * HID * 2);
  __bf16* wt_gcn = (__bf16*)alloc((long long)NLAY * HID * HID * 2);
  __bf16* wt_q   = (__bf16*)alloc((long long)NLAY * HID * HID * 2);
  __bf16* wt_k   = (__bf16*)alloc((long long)NLAY * HID * HID * 2);
  __bf16* wt_v   = (__bf16*)alloc((long long)NLAY * HID * HID * 2);
  __bf16* wt_o   = (__bf16*)alloc((long long)NLAY * HID * HID * 2);
  __bf16* wt_m1  = (__bf16*)alloc((long long)NLAY * HID * 2 * HID * 2);
  __bf16* wt_m2  = (__bf16*)alloc((long long)NLAY * 2 * HID * HID * 2);
  int*    buf_knn= (int*)alloc((long long)BATCH * NNODE * KNN * 4);

  const dim3 blk(256);
  const dim3 gHid(HID / 128, NNODE / 64, BATCH);    // N=384 GEMMs
  const dim3 gMlp(768 / 128, NNODE / 64, BATCH);    // N=768 GEMM
  const dim3 gLn(NNODE / 8, BATCH);

  // 0) one-time bf16 conversions (weights transposed to [N][K])
  cvt_x_kernel<<<dim3((BATCH * NNODE * NCH + 255) / 256), blk, 0, stream>>>(x, buf_xb);
  cvt_w_kernel<<<dim3((NCH * HID + 255) / 256, 1), blk, 0, stream>>>(W_enc, wt_enc, NCH, HID);
  cvt_w_kernel<<<dim3((HID * HID + 255) / 256, NLAY), blk, 0, stream>>>(gcn_W, wt_gcn, HID, HID);
  cvt_w_kernel<<<dim3((HID * HID + 255) / 256, NLAY), blk, 0, stream>>>(Wq, wt_q, HID, HID);
  cvt_w_kernel<<<dim3((HID * HID + 255) / 256, NLAY), blk, 0, stream>>>(Wk, wt_k, HID, HID);
  cvt_w_kernel<<<dim3((HID * HID + 255) / 256, NLAY), blk, 0, stream>>>(Wv, wt_v, HID, HID);
  cvt_w_kernel<<<dim3((HID * HID + 255) / 256, NLAY), blk, 0, stream>>>(Wo, wt_o, HID, HID);
  cvt_w_kernel<<<dim3((HID * 2 * HID + 255) / 256, NLAY), blk, 0, stream>>>(mlp_W1, wt_m1, HID, 2 * HID);
  cvt_w_kernel<<<dim3((2 * HID * HID + 255) / 256, NLAY), blk, 0, stream>>>(mlp_W2, wt_m2, 2 * HID, HID);

  // 1) node encoder: h0 = xb @ W_enc(+b) ; dual f32+bf16 output
  gemm_bf16_wmma<<<gHid, blk, 0, stream>>>(
      buf_xb, (long long)NNODE * NCH, wt_enc, b_enc,
      buf_h, buf_hb, HBN, NNODE, HID, NCH, 0);

  // 2) cosine normalization + kNN hypergraph
  normalize_kernel<<<dim3(NNODE, BATCH), 128, 0, stream>>>(buf_h, buf_nh);
  simtopk_kernel<<<dim3(NNODE / 16, BATCH), blk, 0, stream>>>(buf_nh, buf_knn);

  // 3) GPS layers
  for (int l = 0; l < NLAY; ++l) {
    const __bf16* gW = wt_gcn + (long long)l * HID * HID;
    const __bf16* wq = wt_q   + (long long)l * HID * HID;
    const __bf16* wk = wt_k   + (long long)l * HID * HID;
    const __bf16* wv = wt_v   + (long long)l * HID * HID;
    const __bf16* wo = wt_o   + (long long)l * HID * HID;
    const __bf16* w1 = wt_m1  + (long long)l * HID * 2 * HID;
    const __bf16* w2 = wt_m2  + (long long)l * 2 * HID * HID;
    const float* ls0 = ln_sc + (long long)(l * 3 + 0) * HID;
    const float* lb0 = ln_bi + (long long)(l * 3 + 0) * HID;
    const float* ls1 = ln_sc + (long long)(l * 3 + 1) * HID;
    const float* lb1 = ln_bi + (long long)(l * 3 + 1) * HID;
    const float* ls2 = ln_sc + (long long)(l * 3 + 2) * HID;
    const float* lb2 = ln_bi + (long long)(l * 3 + 2) * HID;

    // local GCN: xw = hcur @ gcn_W + b ; h_local = LN(hcur + agg)
    gemm_bf16_wmma<<<gHid, blk, 0, stream>>>(buf_hb, HBN, gW, gcn_b + l * HID,
                                             buf_xw, nullptr, HBN, NNODE, HID, HID, 0);
    gcn_ln_kernel<<<gLn, blk, 0, stream>>>(buf_h, buf_xw, buf_knn, ls0, lb0, buf_hl);

    // global attention (q/k/v are bf16-only outputs)
    gemm_bf16_wmma<<<gHid, blk, 0, stream>>>(buf_hb, HBN, wq, bq + l * HID,
                                             nullptr, buf_qb, HBN, NNODE, HID, HID, 0);
    gemm_bf16_wmma<<<gHid, blk, 0, stream>>>(buf_hb, HBN, wk, bk + l * HID,
                                             nullptr, buf_kb, HBN, NNODE, HID, HID, 0);
    gemm_bf16_wmma<<<gHid, blk, 0, stream>>>(buf_hb, HBN, wv, bv + l * HID,
                                             nullptr, buf_vb, HBN, NNODE, HID, HID, 0);
    attn_kernel<<<dim3(NNODE / 16, NHEAD, BATCH), blk, 0, stream>>>(buf_qb, buf_kb, buf_vb, buf_ob);
    gemm_bf16_wmma<<<gHid, blk, 0, stream>>>(buf_ob, HBN, wo, bo + l * HID,
                                             buf_xw, nullptr, HBN, NNODE, HID, HID, 0);
    ln_kernel<<<gLn, blk, 0, stream>>>(buf_h, buf_xw, ls1, lb1, buf_ha, nullptr);

    // combine + FFN
    add_kernel<<<dim3((unsigned)(SZ / 256)), blk, 0, stream>>>(buf_hl, buf_ha, buf_cm,
                                                               buf_cb, (int)SZ);
    gemm_bf16_wmma<<<gMlp, blk, 0, stream>>>(buf_cb, HBN, w1, mlp_b1 + l * 2 * HID,
                                             nullptr, buf_t1, H2, NNODE, 2 * HID, HID, 1);
    gemm_bf16_wmma<<<gHid, blk, 0, stream>>>(buf_t1, H2, w2, mlp_b2 + l * HID,
                                             buf_xw, nullptr, HBN, NNODE, HID, 2 * HID, 0);
    ln_kernel<<<gLn, blk, 0, stream>>>(buf_cm, buf_xw, ls2, lb2, buf_h, buf_hb);
  }

  // 4) pool + head
  pool_kernel<<<dim3(BATCH), HID, 0, stream>>>(buf_h, buf_pool);
  head1_kernel<<<dim3(BATCH), HID, 0, stream>>>(buf_pool, lin1_W, lin1_b, buf_z);
  head2_kernel<<<dim3(BATCH), NOUT, 0, stream>>>(buf_z, lin2_W, lin2_b, (float*)d_out);
}
